// ODConv_53034256171262
// MI455X (gfx1250) — matrix operand order repaired and checked by reading, not compiled, not run on hardware
//
#include <hip/hip_runtime.h>

#define NB   16
#define KK   4
#define CIN  256
#define COUT 256
#define HH   56
#define WW   56
#define RED  16
#define PIX  (HH*WW)    // 3136
#define XP   58         // padded spatial
#define TAPS 9

typedef __attribute__((ext_vector_type(16))) __bf16        v16bf;
typedef __attribute__((ext_vector_type(8)))  float         v8f;
typedef __attribute__((ext_vector_type(4)))  unsigned int  u32x4;

union FragBF { v16bf v; u32x4 u[2]; };

__device__ __forceinline__ unsigned short f2bf(float f) {
  unsigned u = __float_as_uint(f);
  u += 0x7FFFu + ((u >> 16) & 1u);          // round to nearest even
  return (unsigned short)(u >> 16);
}

// LDS byte address (wave-relative) from a generic pointer to a __shared__ object.
__device__ __forceinline__ unsigned ldsoff(const void* p) {
  return (unsigned)(unsigned long long)(const __attribute__((address_space(3))) void*)p;
}

// Async global->LDS 16B copy (GVS mode: SGPR64 base + per-lane 32-bit byte offset).
// Tracked by ASYNCcnt; completion enforced with s_wait_asynccnt below.
__device__ __forceinline__ void async_b128(unsigned lds_addr, const unsigned short* base,
                                           unsigned byte_off) {
  asm volatile("global_load_async_to_lds_b128 %0, %1, %2"
               :: "v"(lds_addr), "v"(byte_off), "s"(base)
               : "memory");
}

// ---------------- global average pool: gap[n][c] ----------------
__global__ void k_gap(const float* __restrict__ x, float* __restrict__ gap) {
  const int c = blockIdx.x, n = blockIdx.y, t = threadIdx.x;
  const float* p = x + (size_t)(n * CIN + c) * PIX;
  float s = 0.f;
  for (int i = t; i < PIX; i += 256) s += p[i];
  __shared__ float red[256];
  red[t] = s; __syncthreads();
  for (int off = 128; off > 0; off >>= 1) {
    if (t < off) red[t] += red[t + off];
    __syncthreads();
  }
  if (t == 0) gap[n * CIN + c] = red[0] * (1.f / PIX);
}

// ---------------- attention trunk + 4 branches ----------------
__global__ void k_attn(const float* __restrict__ gap,
                       const float* __restrict__ attn_w, const float* __restrict__ attn_b,
                       const float* __restrict__ fc1_w,  const float* __restrict__ fc1_b,
                       const float* __restrict__ fc2_w,  const float* __restrict__ fc2_b,
                       const float* __restrict__ fc3_w,  const float* __restrict__ fc3_b,
                       const float* __restrict__ fc4_w,  const float* __restrict__ fc4_b,
                       const float* __restrict__ mtx,
                       float* __restrict__ att1, float* __restrict__ att2,
                       float* __restrict__ att3, float* __restrict__ m4) {
  const int n = blockIdx.x, t = threadIdx.x;
  __shared__ float g[CIN];
  __shared__ float a[RED];
  __shared__ float l4[KK];
  g[t] = gap[n * CIN + t];
  __syncthreads();
  if (t < RED) {
    float s = attn_b[t];
    for (int c = 0; c < CIN; ++c) s += attn_w[t * CIN + c] * g[c];
    a[t] = fmaxf(s, 0.f);
  }
  __syncthreads();
  if (t < TAPS) {
    float s = fc1_b[t];
    for (int j = 0; j < RED; ++j) s += fc1_w[t * RED + j] * a[j];
    att1[n * TAPS + t] = 1.f / (1.f + __expf(-s));
  }
  {
    float s2 = fc2_b[t], s3 = fc3_b[t];
    for (int j = 0; j < RED; ++j) {
      s2 += fc2_w[t * RED + j] * a[j];
      s3 += fc3_w[t * RED + j] * a[j];
    }
    att2[n * CIN + t]  = 1.f / (1.f + __expf(-s2));
    att3[n * COUT + t] = 1.f / (1.f + __expf(-s3));
  }
  if (t < KK) {
    float s = fc4_b[t];
    for (int j = 0; j < RED; ++j) s += fc4_w[t * RED + j] * a[j];
    l4[t] = s;
  }
  __syncthreads();
  if (t == 0) {
    float mx = fmaxf(fmaxf(l4[0], l4[1]), fmaxf(l4[2], l4[3]));
    float e0 = __expf(l4[0] - mx), e1 = __expf(l4[1] - mx);
    float e2 = __expf(l4[2] - mx), e3 = __expf(l4[3] - mx);
    float inv = 1.f / (e0 + e1 + e2 + e3);
    m4[n * KK + 0] = e0 * inv * mtx[0];
    m4[n * KK + 1] = e1 * inv * mtx[1];
    m4[n * KK + 2] = e2 * inv * mtx[2];
    m4[n * KK + 3] = e3 * inv * mtx[3];
  }
}

// ---------------- zero scratch (padded image halo) ----------------
__global__ void k_zero(unsigned int* __restrict__ p, size_t nwords) {
  size_t i = (size_t)blockIdx.x * blockDim.x + threadIdx.x;
  const size_t stride = (size_t)gridDim.x * blockDim.x;
  for (; i < nwords; i += stride) p[i] = 0u;
}

// -------- pad + transpose to channels-last bf16: xp[n][y][x][c] --------
__global__ void k_pad(const float* __restrict__ x, unsigned short* __restrict__ xp) {
  const int pt = blockIdx.x, ct = blockIdx.y, n = blockIdx.z, t = threadIdx.x;
  const int p0 = pt * 32, c0 = ct * 32;
  __shared__ float tile[32][33];
  {
    const int pl = t & 31, cl0 = t >> 5;
    for (int j = 0; j < 4; ++j) {
      const int cl = cl0 + j * 8;
      tile[cl][pl] = x[(size_t)(n * CIN + c0 + cl) * PIX + p0 + pl];
    }
  }
  __syncthreads();
  {
    const int cl = t & 31, pl0 = t >> 5;
    for (int j = 0; j < 4; ++j) {
      const int pl = pl0 + j * 8;
      const int p = p0 + pl;
      const int y = p / WW, xw = p % WW;
      xp[(((size_t)n * XP + (y + 1)) * XP + (xw + 1)) * CIN + c0 + cl] = f2bf(tile[cl][pl]);
    }
  }
}

// -------- fused dynamic weight: wf[n][o][tap][i] (bf16) --------
__global__ void k_wfuse(const float* __restrict__ weight, const float* __restrict__ m4,
                        const float* __restrict__ att1, const float* __restrict__ att2,
                        const float* __restrict__ att3, unsigned short* __restrict__ wf) {
  const int o = blockIdx.x, n = blockIdx.y, t = threadIdx.x;
  __shared__ float acc[CIN * TAPS];          // 2304 floats, layout [i][tap]
  __shared__ float a1[TAPS];
  __shared__ float m4s[KK];
  if (t < KK)   m4s[t] = m4[n * KK + t];
  if (t < TAPS) a1[t]  = att1[n * TAPS + t];
  const float a2v = att2[n * CIN + t];
  const float a3  = att3[n * COUT + o];
  __syncthreads();
  const size_t wbase   = ((size_t)(n * KK) * COUT + o) * (CIN * TAPS);
  const size_t kstride = (size_t)COUT * CIN * TAPS;
  for (int j = 0; j < TAPS; ++j) {
    const int idx = t + j * 256;
    float s = 0.f;
    #pragma unroll
    for (int k = 0; k < KK; ++k) s += m4s[k] * weight[wbase + (size_t)k * kstride + idx];
    acc[idx] = s;
  }
  __syncthreads();
  const size_t obase = (size_t)(n * COUT + o) * TAPS * CIN;
  for (int j = 0; j < TAPS; ++j) {
    const float v = acc[t * TAPS + j] * a3 * a1[j] * a2v;
    wf[obase + (size_t)j * CIN + t] = f2bf(v);
  }
}

// -------- implicit-GEMM conv: bf16 WMMA, async-to-LDS triple buffer --------
// C[n][o][p] = sum_{tap,i} wf[n][o][tap][i] * xp[n][y+kh][x+kw][i]
// Block tile 128(M=o) x 128(P=pixels), Kblk=32 (72 steps), 8 waves (2x4),
// 8 accumulators per wave. One barrier per K-step.
__global__ void __launch_bounds__(256) k_gemm(
    const unsigned short* __restrict__ wf, const unsigned short* __restrict__ xp,
    const float* __restrict__ bn_g, const float* __restrict__ bn_b,
    const float* __restrict__ bn_m, const float* __restrict__ bn_v,
    float* __restrict__ out) {
  const int p0 = blockIdx.x * 128;
  const int m0 = blockIdx.y * 128;
  const int n  = blockIdx.z;
  const int t  = threadIdx.x;

  __shared__ __align__(16) unsigned short Als[3][128 * 32];  // [buf][row][k]
  __shared__ __align__(16) unsigned short Bls[3][128 * 32];  // [buf][pcol][k]
  __shared__ float bns[128], bnbias[128];

  if (t < 128) {
    const int o = m0 + t;
    const float s = bn_g[o] * rsqrtf(bn_v[o] + 1e-5f);
    bns[t]    = s;
    bnbias[t] = bn_b[o] - bn_m[o] * s;
  }

  const int wave = t >> 5, lane = t & 31;
  const int wm = wave & 1;        // M half (64 rows)
  const int wp = wave >> 1;       // P quarter (32 cols)
  v8f acc[4][2] = {};

  // per-thread tile-load mapping (constant over K-steps)
  const int arow = t >> 2, aq = t & 3;      // A: 2 threads*? -> 4 chunks/row, rows arow & arow+64
  const int pl = t >> 1, half = t & 1;      // B: 2 threads per pixel, 16ch each
  int p = p0 + pl; if (p > PIX - 1) p = PIX - 1;   // clamp OOB loads
  const int py = p / WW, px = p % WW;

  const unsigned alds = ldsoff(&Als[0][0]);
  const unsigned blds = ldsoff(&Bls[0][0]);
  const unsigned aoff = (unsigned)(arow * 32 + aq * 8) * 2;       // bytes within A buffer
  const unsigned boff = (unsigned)(pl * 32 + half * 16) * 2;      // bytes within B buffer
  // global byte offsets, minus the per-K-step part
  const unsigned abase0 = (unsigned)((n * COUT + m0 + arow) * TAPS * CIN) * 2 + (unsigned)aq * 16;
  const unsigned abase1 = abase0 + (unsigned)(64 * TAPS * CIN) * 2;
  const unsigned bbase  = (unsigned)(((n * XP + py) * XP + px) * CIN) * 2 + (unsigned)half * 32;

  auto issue = [&](int kb) {
    const int tap = kb >> 3;                  // 0..8
    const int i0  = (kb & 7) * 32;            // channel block
    const unsigned ib   = (unsigned)(kb % 3) * 8192u;   // LDS buffer stride in bytes
    const unsigned addA = (unsigned)(tap * CIN + i0) * 2;
    const unsigned addB = (unsigned)(((tap / 3) * XP + (tap % 3)) * CIN + i0) * 2;
    async_b128(alds + ib + aoff,        wf, abase0 + addA);
    async_b128(alds + ib + aoff + 4096, wf, abase1 + addA);
    async_b128(blds + ib + boff,        xp, bbase + addB);
    async_b128(blds + ib + boff + 16,   xp, bbase + addB + 16);
  };

  issue(0);
  for (int kb = 0; kb < 72; ++kb) {
    if (kb < 71) {
      issue(kb + 1);
      // 4 newest outstanding ops are tile kb+1's -> tile kb fully in LDS
      asm volatile("s_wait_asynccnt 0x4" ::: "memory");
    } else {
      asm volatile("s_wait_asynccnt 0x0" ::: "memory");
    }
    __syncthreads();

    const unsigned short* As = Als[kb % 3];
    const unsigned short* Bs = Bls[kb % 3];
    FragBF a[4], b[2];
    const int am = lane & 15, akb = (lane >> 4) * 8;     // A: K halves 0-7 / 8-15 (+16)
    #pragma unroll
    for (int mt = 0; mt < 4; ++mt) {
      const unsigned short* src = &As[(wm * 64 + mt * 16 + am) * 32 + akb];
      a[mt].u[0] = *(const u32x4*)src;          // K = akb .. akb+7
      a[mt].u[1] = *(const u32x4*)(src + 16);   // K = akb+16 .. akb+23
    }
    const int bncol = lane & 15, bkb = (lane >> 4) * 8;  // B: K halves 0-7 / 8-15 (+16)
    #pragma unroll
    for (int pt = 0; pt < 2; ++pt) {
      const unsigned short* src = &Bs[(wp * 32 + pt * 16 + bncol) * 32 + bkb];
      b[pt].u[0] = *(const u32x4*)src;
      b[pt].u[1] = *(const u32x4*)(src + 16);
    }
    #pragma unroll
    for (int mt = 0; mt < 4; ++mt)
      #pragma unroll
      for (int pt = 0; pt < 2; ++pt)
        acc[mt][pt] = __builtin_amdgcn_wmma_f32_16x16x32_bf16(
            false, a[mt].v, false, b[pt].v, (short)0, acc[mt][pt], false, false);
  }

  // --- epilogue: BN + ReLU, guarded stores ---
  #pragma unroll
  for (int mt = 0; mt < 4; ++mt) {
    const int mlbase = wm * 64 + mt * 16 + ((lane >> 4) * 8);
    #pragma unroll
    for (int pt = 0; pt < 2; ++pt) {
      const int pcol = p0 + wp * 32 + pt * 16 + (lane & 15);
      if (pcol < PIX) {
        #pragma unroll
        for (int g = 0; g < 8; ++g) {
          const int ml = mlbase + g;
          const float v = acc[mt][pt][g] * bns[ml] + bnbias[ml];
          out[(size_t)(n * COUT + m0 + ml) * PIX + pcol] = fmaxf(v, 0.f);
        }
      }
    }
  }
}

extern "C" void kernel_launch(void* const* d_in, const int* in_sizes, int n_in,
                              void* d_out, int out_size, void* d_ws, size_t ws_size,
                              hipStream_t stream) {
  const float* x      = (const float*)d_in[0];
  const float* weight = (const float*)d_in[1];
  const float* mtx    = (const float*)d_in[2];
  const float* attn_w = (const float*)d_in[3];
  const float* attn_b = (const float*)d_in[4];
  const float* fc1_w  = (const float*)d_in[5];
  const float* fc1_b  = (const float*)d_in[6];
  const float* fc2_w  = (const float*)d_in[7];
  const float* fc2_b  = (const float*)d_in[8];
  const float* fc3_w  = (const float*)d_in[9];
  const float* fc3_b  = (const float*)d_in[10];
  const float* fc4_w  = (const float*)d_in[11];
  const float* fc4_b  = (const float*)d_in[12];
  const float* bn_g   = (const float*)d_in[13];
  const float* bn_b   = (const float*)d_in[14];
  const float* bn_m   = (const float*)d_in[15];
  const float* bn_v   = (const float*)d_in[16];

  char* ws = (char*)d_ws;
  float* gap  = (float*)(ws + 0);          // 16*256 f32
  float* att1 = (float*)(ws + 16384);      // 16*9   f32
  float* att2 = (float*)(ws + 17408);      // 16*256 f32
  float* att3 = (float*)(ws + 33792);      // 16*256 f32
  float* m4   = (float*)(ws + 50176);      // 16*4   f32 (softmax*mtx)
  unsigned short* wf = (unsigned short*)(ws + 51200);                       // 16*256*9*256 bf16
  unsigned short* xp = (unsigned short*)(ws + 51200 + (size_t)NB * COUT * TAPS * CIN * 2); // 16*58*58*256 bf16

  k_gap <<<dim3(CIN, NB), 256, 0, stream>>>(x, gap);
  k_attn<<<NB, 256, 0, stream>>>(gap, attn_w, attn_b, fc1_w, fc1_b, fc2_w, fc2_b,
                                 fc3_w, fc3_b, fc4_w, fc4_b, mtx, att1, att2, att3, m4);
  const size_t xpwords = (size_t)NB * XP * XP * CIN * 2 / 4;
  k_zero<<<2048, 256, 0, stream>>>((unsigned int*)xp, xpwords);
  k_pad <<<dim3(PIX / 32, CIN / 32, NB), 256, 0, stream>>>(x, xp);
  k_wfuse<<<dim3(COUT, NB), 256, 0, stream>>>(weight, m4, att1, att2, att3, wf);
  k_gemm<<<dim3((PIX + 127) / 128, COUT / 128, NB), 256, 0, stream>>>(
      wf, xp, bn_g, bn_b, bn_m, bn_v, (float*)d_out);
}
